// DiTBlock_35665408426665
// MI455X (gfx1250) — matrix operand order repaired and checked by reading, not compiled, not run on hardware
//
#include <hip/hip_runtime.h>
#include <hip/hip_bf16.h>
#include <math.h>
#include <stdint.h>

typedef __attribute__((ext_vector_type(16))) _Float16 v16h;
typedef __attribute__((ext_vector_type(8)))  float    v8f;
typedef __attribute__((ext_vector_type(4)))  unsigned int u32x4;
typedef __attribute__((ext_vector_type(8)))  int      i32x8;
typedef __attribute__((ext_vector_type(4)))  int      i32x4;

#define DMODEL 1024
#define NHEAD  16
#define HDIM   64
#define FFDIM  4096
#define SEQ    1024
#define BATCH  8
#define TOKENS (BATCH * SEQ)

#define KCHUNK 64            // K elements staged per TDM transfer
#define LDSROW 72            // KCHUNK + 8 halves (TDM pad: +16B per 128B row)

// ---------------- WMMA helpers (CDNA5 16x16x32 f16 -> f32) ----------------

__device__ __forceinline__ v8f wmma16(v16h a, v16h b, v8f c) {
  return __builtin_amdgcn_wmma_f32_16x16x32_f16(false, a, false, b, (short)0, c,
                                                false, false);
}

// A-matrix 16x32 f16 fragment from row-major A (lda in elements).
// lane 0-15: M=lane, K = kb+e (e=0..15)
// lane16-31: M=lane-16, K = kb+16+e (e=0..15)
__device__ __forceinline__ v16h load_a_frag(const _Float16* __restrict__ A,
                                            int lda, int row0, int kb, int lane) {
  int row = row0 + (lane & 15);
  int khi = (lane >> 4) << 4;
  const _Float16* p = A + (size_t)row * lda + kb + khi;
  union { float4 f[2]; v16h h; } u;
  u.f[0] = *reinterpret_cast<const float4*>(p);        // K = kb+khi .. +7
  u.f[1] = *reinterpret_cast<const float4*>(p + 8);    // K = kb+khi+8 .. +15
  return u.h;
}

// B-matrix 32x16 f16 fragment from Bt[N][K] (row-major, ldb in elements).
// lane 0-15: N=lane,  K = kb+e (e=0..15); lane16-31: N=lane-16, K = kb+16+e
__device__ __forceinline__ v16h load_b_frag(const _Float16* __restrict__ Bt,
                                            int ldb, int n0, int kb, int lane) {
  int n = n0 + (lane & 15);
  int koff = (lane >> 4) << 4;
  const _Float16* p = Bt + (size_t)n * ldb + kb + koff;
  union { float4 f[2]; v16h h; } u;
  u.f[0] = *reinterpret_cast<const float4*>(p);
  u.f[1] = *reinterpret_cast<const float4*>(p + 8);
  return u.h;
}

// B fragment from the TDM-staged LDS tile (row pitch LDSROW halves).
__device__ __forceinline__ v16h lds_b_frag(const _Float16* sm, int n0, int kk,
                                           int lane) {
  int n = n0 + (lane & 15);
  int koff = (lane >> 4) << 4;
  const _Float16* p = sm + n * LDSROW + kk + koff;
  union { float4 f[2]; v16h h; } u;
  u.f[0] = *reinterpret_cast<const float4*>(p);
  u.f[1] = *reinterpret_cast<const float4*>(p + 8);
  return u.h;
}

// ---------------- Tensor Data Mover: stage 128 x KCHUNK f16 tile ----------
// D# per ISA ch8: group0 = {count/lds_addr/global_addr/type}, group1 =
// {data_size, pad, tensor dims/strides, tile dims}. 2D tensor -> groups 2/3
// are zero. LDS pad: interval 2<<4=32 DW (=128B row), amount 3+1=4 DW (16B).
__device__ __forceinline__ void tdm_load_tile_b(const _Float16* gptr,
                                                unsigned lds_byte_off, int ldK) {
  unsigned long long ga = (unsigned long long)(uintptr_t)gptr;
  u32x4 g0;
  g0[0] = 1u;                                            // count=1, user mode
  g0[1] = lds_byte_off;                                  // lds_addr
  g0[2] = (unsigned)ga;                                  // global_addr lo
  g0[3] = (unsigned)((ga >> 32) & 0x01ffffffu) | (2u << 30);  // addr hi | type=2
  i32x8 g1;
  g1[0] = (1 << 16) | (1 << 20) | (4 << 22) | (3 << 25); // 2B elems, pad cfg
  g1[1] = (KCHUNK & 0xffff) << 16;                       // tensor_dim0 lo16
  g1[2] = (KCHUNK >> 16) | (128 << 16);                  // dim0 hi | dim1 lo=128
  g1[3] = (KCHUNK << 16);                                // dim1 hi=0 | tile_dim0
  g1[4] = 128;                                           // tile_dim1 | tile_dim2=0
  g1[5] = ldK;                                           // tensor_dim0_stride lo
  g1[6] = 0;                                             // stride0 hi | stride1 lo
  g1[7] = 0;                                             // stride1 hi
  i32x4 z4 = {0, 0, 0, 0};
#if defined(__clang_major__) && __clang_major__ >= 23
  i32x8 z8 = {0, 0, 0, 0, 0, 0, 0, 0};
  __builtin_amdgcn_tensor_load_to_lds(g0, g1, z4, z4, z8, 0);
#else
  __builtin_amdgcn_tensor_load_to_lds(g0, g1, z4, z4, 0);
#endif
}

// ---------------- Prep kernels ----------------

__global__ __launch_bounds__(256) void transpose_f16_kernel(
    const float* __restrict__ W, _Float16* __restrict__ Wt, int K, int N) {
  int idx = blockIdx.x * 256 + threadIdx.x;
  if (idx >= K * N) return;
  int n = idx / K;
  int k = idx - n * K;
  Wt[idx] = (_Float16)W[(size_t)k * N + n];
}

__global__ __launch_bounds__(256) void ada_kernel(
    const float* __restrict__ c, const float* __restrict__ w,
    const float* __restrict__ bias, float* __restrict__ ssg) {
  int idx = blockIdx.x * 256 + threadIdx.x;   // < BATCH * 6144
  int b = idx / 6144, n = idx - b * 6144;
  float acc = bias[n];
  for (int d = 0; d < DMODEL; ++d) {
    float cv = c[b * DMODEL + d];
    float s = cv / (1.f + expf(-cv));
    acc = fmaf(s, w[(size_t)d * 6144 + n], acc);
  }
  ssg[idx] = acc;
}

__global__ __launch_bounds__(256) void ln_mod_kernel(
    const float* __restrict__ x, const float* __restrict__ lw,
    const float* __restrict__ lb, const float* __restrict__ ssg,
    int shift_off, int scale_off, _Float16* __restrict__ out) {
  __shared__ float s_sum[256], s_sq[256];
  const int row = blockIdx.x;           // 0..TOKENS-1
  const int b = row >> 10;
  const float* xr = x + (size_t)row * DMODEL;
  float sum = 0.f, sq = 0.f;
  for (int j = threadIdx.x; j < DMODEL; j += 256) {
    float v = xr[j]; sum += v; sq += v * v;
  }
  s_sum[threadIdx.x] = sum; s_sq[threadIdx.x] = sq;
  __syncthreads();
  for (int st = 128; st > 0; st >>= 1) {
    if ((int)threadIdx.x < st) {
      s_sum[threadIdx.x] += s_sum[threadIdx.x + st];
      s_sq[threadIdx.x]  += s_sq[threadIdx.x + st];
    }
    __syncthreads();
  }
  const float mu = s_sum[0] * (1.f / DMODEL);
  const float var = s_sq[0] * (1.f / DMODEL) - mu * mu;
  const float rstd = rsqrtf(var + 1e-5f);
  const float* shiftv = ssg + (size_t)b * 6144 + shift_off;
  const float* scalev = ssg + (size_t)b * 6144 + scale_off;
  for (int j = threadIdx.x; j < DMODEL; j += 256) {
    float v = (xr[j] - mu) * rstd * lw[j] + lb[j];
    v = v * (1.f + scalev[j]) + shiftv[j];
    out[(size_t)row * DMODEL + j] = (_Float16)v;
  }
}

// ---------------- WMMA GEMM, TDM-staged B panel, fused epilogues ----------
// C(M x N) = A(M x K, f16 row-major) * Bt(N x K, f16 row-major)^T
// block tile 64(M) x 128(N); 8 waves, each 32x32 (2x2 WMMA tiles).
// B weight panel staged to LDS by the Tensor Data Mover, double-buffered.
template <int MODE>
__global__ __launch_bounds__(256) void gemm_kernel(
    const _Float16* __restrict__ A, const _Float16* __restrict__ Bt,
    const float* __restrict__ bias, const float* __restrict__ ssg,
    const float* __restrict__ resid, float* __restrict__ outf,
    _Float16* __restrict__ outh, _Float16* __restrict__ qb,
    _Float16* __restrict__ kbuf, _Float16* __restrict__ vt, int N, int K) {
  __shared__ _Float16 bsm[2][128 * LDSROW];
  const int lane  = threadIdx.x & 31;
  const int wid   = threadIdx.x >> 5;
  const int mbase = blockIdx.y * 64 + (wid >> 2) * 32;
  const int nblock = blockIdx.x * 128;
  const int nloc  = (wid & 3) * 32;

  const unsigned lds0 = (unsigned)(uintptr_t)&bsm[0][0];
  const unsigned lds1 = (unsigned)(uintptr_t)&bsm[1][0];

  if (wid == 0) {
    tdm_load_tile_b(Bt + (size_t)nblock * K, lds0, K);
    __builtin_amdgcn_s_wait_tensorcnt(0);
  }
  __syncthreads();

  v8f acc[2][2] = {};
  int buf = 0;
  for (int kc = 0; kc < K; kc += KCHUNK, buf ^= 1) {
    if (wid == 0 && kc + KCHUNK < K)    // prefetch next weight panel chunk
      tdm_load_tile_b(Bt + (size_t)nblock * K + kc + KCHUNK,
                      buf ? lds0 : lds1, K);
    for (int kk = 0; kk < KCHUNK; kk += 32) {
      v16h a0 = load_a_frag(A, K, mbase,      kc + kk, lane);
      v16h a1 = load_a_frag(A, K, mbase + 16, kc + kk, lane);
      v16h b0 = lds_b_frag(&bsm[buf][0], nloc,      kk, lane);
      v16h b1 = lds_b_frag(&bsm[buf][0], nloc + 16, kk, lane);
      acc[0][0] = wmma16(a0, b0, acc[0][0]);
      acc[0][1] = wmma16(a0, b1, acc[0][1]);
      acc[1][0] = wmma16(a1, b0, acc[1][0]);
      acc[1][1] = wmma16(a1, b1, acc[1][1]);
    }
    if (wid == 0) __builtin_amdgcn_s_wait_tensorcnt(0);
    __syncthreads();
  }

  const int mhi = (lane >> 4) << 3;   // C layout: lane = N col, VGPR r = M row
  const int nlo = lane & 15;
  for (int i = 0; i < 2; ++i)
    for (int j = 0; j < 2; ++j) {
      const int n = nblock + nloc + j * 16 + nlo;
      const float bv = bias[n];
      for (int r = 0; r < 8; ++r) {
        const int m = mbase + i * 16 + mhi + r;
        float v = acc[i][j][r] + bv;
        if constexpr (MODE == 0) {
          const int b = m >> 10, l = m & 1023;
          if (n < 1024) {
            qb[(((size_t)b * NHEAD + (n >> 6)) * SEQ + l) * HDIM + (n & 63)] =
                (_Float16)v;
          } else if (n < 2048) {
            const int n2 = n - 1024;
            kbuf[(((size_t)b * NHEAD + (n2 >> 6)) * SEQ + l) * HDIM + (n2 & 63)] =
                (_Float16)v;
          } else {
            const int n2 = n - 2048;
            vt[(((size_t)b * NHEAD + (n2 >> 6)) * HDIM + (n2 & 63)) * SEQ + l] =
                (_Float16)v;
          }
        } else if constexpr (MODE == 1) {
          const size_t o = (size_t)m * DMODEL + n;
          const float g = ssg[(size_t)(m >> 10) * 6144 + 2048 + n];
          outf[o] = resid[o] + g * v;
        } else if constexpr (MODE == 2) {
          const float t = v;
          const float gl =
              0.5f * t * (1.f + tanhf(0.7978845608f * (t + 0.044715f * t * t * t)));
          outh[(size_t)m * FFDIM + n] = (_Float16)gl;
        } else {
          const size_t o = (size_t)m * DMODEL + n;
          const float g = ssg[(size_t)(m >> 10) * 6144 + 5120 + n];
          outf[o] = resid[o] + g * v;
        }
      }
    }
}

// ---------------- Flash attention (one wave / 16 queries) ----------------
// Computes S^T = K_tile(16k x 64) * Q^T(64 x 16q) so softmax state is
// per-lane (lane = query). Then O^T += V^T(16d x 32k) * P^T(32k x 16q).
__global__ __launch_bounds__(32) void attn_kernel(
    const _Float16* __restrict__ Q, const _Float16* __restrict__ Kmat,
    const _Float16* __restrict__ Vt, _Float16* __restrict__ Ab) {
  const int lane = threadIdx.x;
  const int qt = blockIdx.x, hh = blockIdx.y, b = blockIdx.z;
  const _Float16* Qbase = Q    + ((size_t)(b * NHEAD + hh) * SEQ + qt * 16) * HDIM;
  const _Float16* Kbase = Kmat + (size_t)(b * NHEAD + hh) * SEQ * HDIM;
  const _Float16* Vbase = Vt   + (size_t)(b * NHEAD + hh) * HDIM * SEQ;

  const v16h bq0 = load_b_frag(Qbase, HDIM, 0, 0, lane);
  const v16h bq1 = load_b_frag(Qbase, HDIM, 0, 32, lane);

  v8f accO[4] = {};              // O^T: 4 chunks of (16 d-rows x 16 queries)
  float mrun = -3.0e38f, lsum = 0.f;
  const float scale = 0.125f;    // 1/sqrt(64)
  const bool lo = lane < 16;

  for (int kb = 0; kb < SEQ; kb += 32) {
    v8f t0 = {}, t1 = {};
    t0 = wmma16(load_a_frag(Kbase, HDIM, kb,      0,  lane), bq0, t0);
    t0 = wmma16(load_a_frag(Kbase, HDIM, kb,      32, lane), bq1, t0);
    t1 = wmma16(load_a_frag(Kbase, HDIM, kb + 16, 0,  lane), bq0, t1);
    t1 = wmma16(load_a_frag(Kbase, HDIM, kb + 16, 32, lane), bq1, t1);

    float s0[8], s1[8], tmax = -3.0e38f;
    for (int r = 0; r < 8; ++r) {
      s0[r] = t0[r] * scale;
      s1[r] = t1[r] * scale;
      tmax = fmaxf(tmax, fmaxf(s0[r], s1[r]));
    }
    tmax = fmaxf(tmax, __shfl_xor(tmax, 16));
    const float newm = fmaxf(mrun, tmax);
    const float corr = expf(mrun - newm);
    float p0[8], p1[8], psum = 0.f;
    for (int r = 0; r < 8; ++r) {
      p0[r] = expf(s0[r] - newm);
      p1[r] = expf(s1[r] - newm);
      psum += p0[r] + p1[r];
    }
    psum += __shfl_xor(psum, 16);
    lsum = lsum * corr + psum;
    mrun = newm;

    v16h pf;       // P^T B-fragment (32 keys x 16 queries)
    for (int r = 0; r < 8; ++r) {
      const float x0 = __shfl_xor(p0[r], 16);
      const float x1 = __shfl_xor(p1[r], 16);
      pf[r]     = (_Float16)(lo ? p0[r] : x1);   // keys kb+r   / kb+16+r
      pf[8 + r] = (_Float16)(lo ? x0 : p1[r]);   // keys kb+8+r / kb+24+r
    }

    for (int c = 0; c < 4; ++c) {
      v16h av = load_a_frag(Vbase, SEQ, c * 16, kb, lane);  // V^T rows = d
      for (int r = 0; r < 8; ++r) accO[c][r] *= corr;
      accO[c] = wmma16(av, pf, accO[c]);
    }
  }

  const float inv = (lsum > 0.f) ? 1.f / lsum : 0.f;
  const int q = lane & 15;
  const int dhi = (lane >> 4) << 3;
  for (int c = 0; c < 4; ++c) {
    union { _Float16 h[8]; float4 f; } u;
    for (int r = 0; r < 8; ++r) u.h[r] = (_Float16)(accO[c][r] * inv);
    const size_t o = ((size_t)(b * SEQ + qt * 16 + q)) * DMODEL +
                     hh * HDIM + c * 16 + dhi;
    *reinterpret_cast<float4*>(Ab + o) = u.f;
  }
}

// ---------------- Host launcher ----------------

extern "C" void kernel_launch(void* const* d_in, const int* in_sizes, int n_in,
                              void* d_out, int out_size, void* d_ws, size_t ws_size,
                              hipStream_t stream) {
  const float* x      = (const float*)d_in[0];
  const float* c      = (const float*)d_in[1];
  const float* ln1_w  = (const float*)d_in[2];
  const float* ln1_b  = (const float*)d_in[3];
  const float* ln2_w  = (const float*)d_in[4];
  const float* ln2_b  = (const float*)d_in[5];
  const float* ada_w  = (const float*)d_in[6];
  const float* ada_b  = (const float*)d_in[7];
  const float* qkv_w  = (const float*)d_in[8];
  const float* qkv_b  = (const float*)d_in[9];
  const float* proj_w = (const float*)d_in[10];
  const float* proj_b = (const float*)d_in[11];
  const float* fc1_w  = (const float*)d_in[12];
  const float* fc1_b  = (const float*)d_in[13];
  const float* fc2_w  = (const float*)d_in[14];
  const float* fc2_b  = (const float*)d_in[15];
  float* out = (float*)d_out;

  char* wsb = (char*)d_ws;
  size_t off = 0;
  auto take = [&](size_t bytes) {
    char* p = wsb + off;
    off += (bytes + 255) & ~(size_t)255;
    return (void*)p;
  };
  _Float16* qkvT = (_Float16*)take((size_t)3072 * 1024 * 2);
  _Float16* projT = (_Float16*)take((size_t)1024 * 1024 * 2);
  _Float16* fc1T = (_Float16*)take((size_t)4096 * 1024 * 2);
  _Float16* fc2T = (_Float16*)take((size_t)1024 * 4096 * 2);
  float*    ssg  = (float*)take((size_t)BATCH * 6144 * 4);
  _Float16* h1   = (_Float16*)take((size_t)TOKENS * DMODEL * 2);
  _Float16* Qb   = (_Float16*)take((size_t)TOKENS * DMODEL * 2);
  _Float16* Kb   = (_Float16*)take((size_t)TOKENS * DMODEL * 2);
  _Float16* Vtb  = (_Float16*)take((size_t)TOKENS * DMODEL * 2);
  _Float16* Abuf = (_Float16*)take((size_t)TOKENS * DMODEL * 2);
  float*    x1   = (float*)take((size_t)TOKENS * DMODEL * 4);
  _Float16* h2   = (_Float16*)take((size_t)TOKENS * DMODEL * 2);
  _Float16* h3   = (_Float16*)take((size_t)TOKENS * FFDIM * 2);

  transpose_f16_kernel<<<(3072 * 1024) / 256, 256, 0, stream>>>(qkv_w, qkvT, 1024, 3072);
  transpose_f16_kernel<<<(1024 * 1024) / 256, 256, 0, stream>>>(proj_w, projT, 1024, 1024);
  transpose_f16_kernel<<<(4096 * 1024) / 256, 256, 0, stream>>>(fc1_w, fc1T, 1024, 4096);
  transpose_f16_kernel<<<(4096 * 1024) / 256, 256, 0, stream>>>(fc2_w, fc2T, 4096, 1024);

  ada_kernel<<<(BATCH * 6144) / 256, 256, 0, stream>>>(c, ada_w, ada_b, ssg);

  ln_mod_kernel<<<TOKENS, 256, 0, stream>>>(x, ln1_w, ln1_b, ssg, 0, 1024, h1);

  gemm_kernel<0><<<dim3(3072 / 128, TOKENS / 64), 256, 0, stream>>>(
      h1, qkvT, qkv_b, nullptr, nullptr, nullptr, nullptr, Qb, Kb, Vtb, 3072, 1024);

  attn_kernel<<<dim3(SEQ / 16, NHEAD, BATCH), 32, 0, stream>>>(Qb, Kb, Vtb, Abuf);

  gemm_kernel<1><<<dim3(1024 / 128, TOKENS / 64), 256, 0, stream>>>(
      Abuf, projT, proj_b, ssg, x, x1, nullptr, nullptr, nullptr, nullptr, 1024, 1024);

  ln_mod_kernel<<<TOKENS, 256, 0, stream>>>(x1, ln2_w, ln2_b, ssg, 3072, 4096, h2);

  gemm_kernel<2><<<dim3(4096 / 128, TOKENS / 64), 256, 0, stream>>>(
      h2, fc1T, fc1_b, nullptr, nullptr, nullptr, h3, nullptr, nullptr, nullptr, 4096, 1024);

  gemm_kernel<3><<<dim3(1024 / 128, TOKENS / 64), 256, 0, stream>>>(
      h3, fc2T, fc2_b, ssg, x1, out, nullptr, nullptr, nullptr, nullptr, 1024, 4096);
}
